// T3AWrapper_72550587564093
// MI455X (gfx1250) — matrix operand order repaired and checked by reading, not compiled, not run on hardware
//
#include <hip/hip_runtime.h>
#include <hip/hip_bf16.h>
#include <math.h>

// ---------------------------------------------------------------------------
// Types for the CDNA5 WMMA intrinsics
// ---------------------------------------------------------------------------
typedef __bf16 bf16_t;
typedef bf16_t v16bf __attribute__((ext_vector_type(16)));
typedef bf16_t v8bf  __attribute__((ext_vector_type(8)));
typedef bf16_t v4bf  __attribute__((ext_vector_type(4)));
typedef float  v8f   __attribute__((ext_vector_type(8)));

#define ENT_THR 0.6f
#define MSEL    29          // top-M (30) minus the always-selected prototype
#define EPSN    1e-12f

// float -> bf16, round-to-nearest-even (used only in streaming conversion)
__device__ __forceinline__ bf16_t f2bf(float f) {
    union { float f; unsigned u; } v; v.f = f;
    unsigned r = v.u + 0x7FFFu + ((v.u >> 16) & 1u);
    unsigned short h = (unsigned short)(r >> 16);
    return __builtin_bit_cast(bf16_t, h);
}

// ---------------------------------------------------------------------------
// Streaming f32 -> bf16 conversion (memory bound; done once per matrix)
// n4 = element_count / 4
// ---------------------------------------------------------------------------
__global__ void __launch_bounds__(256)
f32_to_bf16_kernel(const float* __restrict__ X, bf16_t* __restrict__ Y, long n4) {
    long i = (long)blockIdx.x * 256 + threadIdx.x;
    long stride = (long)gridDim.x * 256;
    for (; i < n4; i += stride) {
        float4 v = ((const float4*)X)[i];
        v4bf o;
        o[0] = f2bf(v.x); o[1] = f2bf(v.y); o[2] = f2bf(v.z); o[3] = f2bf(v.w);
        ((v4bf*)Y)[i] = o;
    }
}

// ---------------------------------------------------------------------------
// Per-row inverse L2 norm (matches F.normalize eps semantics)
// ---------------------------------------------------------------------------
__global__ void __launch_bounds__(256)
rownorm_inv_kernel(const float* __restrict__ X, float* __restrict__ rinv, int D) {
    __shared__ float sh[256];
    const int row = blockIdx.x;
    const float* xp = X + (size_t)row * D;
    float s = 0.f;
    for (int d = threadIdx.x; d < D; d += 256) { float v = xp[d]; s += v * v; }
    sh[threadIdx.x] = s; __syncthreads();
    for (int k = 128; k > 0; k >>= 1) {
        if (threadIdx.x < k) sh[threadIdx.x] += sh[threadIdx.x + k];
        __syncthreads();
    }
    if (threadIdx.x == 0) rinv[row] = 1.f / fmaxf(sqrtf(sh[0]), EPSN);
}

// ---------------------------------------------------------------------------
// WMMA GEMM: out(MxN) = X(MxD) @ Y(NxD)^T  [+ bias[n]] [* rowscale[m]]
// bf16 inputs, fp32 accumulate.  Each wave computes a 32x32 output block as
// 2x2 tiles of v_wmma_f32_16x16x32_bf16 -> 4 WMMAs per 8 b128 loads per
// K-step.  4 waves per 128-thread block tile along N.
// ---------------------------------------------------------------------------
__device__ __forceinline__ void store_tile(float* __restrict__ out, const v8f& acc,
                                           int mbase, int nbase, int l16, int h,
                                           int Nrows, const float* __restrict__ bias,
                                           const float* __restrict__ rowscale) {
    const int n = nbase + l16;
    if (n >= Nrows) return;
    const float bv = bias ? bias[n] : 0.f;
    #pragma unroll
    for (int r = 0; r < 8; ++r) {
        const int m = mbase + r + 8 * h;       // C layout: M = vgpr + 8*half
        float v = acc[r];
        if (rowscale) v *= rowscale[m];
        out[(size_t)m * Nrows + n] = v + bv;
    }
}

__global__ void __launch_bounds__(128)
wmma_gemm_nt_bf16_kernel(const bf16_t* __restrict__ X, const bf16_t* __restrict__ Y,
                         const float* __restrict__ bias, const float* __restrict__ rowscale,
                         float* __restrict__ out, int Nrows, int D, int ntilesN32) {
    const int wave = threadIdx.x >> 5;
    const int lane = threadIdx.x & 31;
    const int nt   = blockIdx.x * 4 + wave;        // 32-column tile index
    if (nt >= ntilesN32) return;                   // wave-uniform exit
    const int mt  = blockIdx.y;                    // 32-row tile index
    const int m0  = mt * 32, n0 = nt * 32;
    const int l16 = lane & 15, h = lane >> 4;

    // A rows (M is a multiple of 32); B rows clamped at the N tail
    const bf16_t* __restrict__ pa0 = X + (size_t)(m0 + l16)      * D;
    const bf16_t* __restrict__ pa1 = X + (size_t)(m0 + 16 + l16) * D;
    int br0 = n0 + l16;      if (br0 >= Nrows) br0 = Nrows - 1;
    int br1 = n0 + 16 + l16; if (br1 >= Nrows) br1 = Nrows - 1;
    const bf16_t* __restrict__ pb0 = Y + (size_t)br0 * D;
    const bf16_t* __restrict__ pb1 = Y + (size_t)br1 * D;

    v8f acc00 = {}, acc01 = {}, acc10 = {}, acc11 = {};

    for (int kk = 0; kk < D; kk += 32) {
        if (kk + 32 < D) {                         // global_prefetch_b8 on gfx1250
            __builtin_prefetch(pa0 + kk + 32, 0, 1);
            __builtin_prefetch(pb0 + kk + 32, 0, 1);
        }
        // A fragments: lane = row l16; half h -> contiguous K {h*16 .. h*16+15}
        v8bf a0lo = *(const v8bf*)(pa0 + kk + h * 16);
        v8bf a0hi = *(const v8bf*)(pa0 + kk + h * 16 + 8);
        v8bf a1lo = *(const v8bf*)(pa1 + kk + h * 16);
        v8bf a1hi = *(const v8bf*)(pa1 + kk + h * 16 + 8);
        v16bf a0 = __builtin_shufflevector(a0lo, a0hi, 0,1,2,3,4,5,6,7,8,9,10,11,12,13,14,15);
        v16bf a1 = __builtin_shufflevector(a1lo, a1hi, 0,1,2,3,4,5,6,7,8,9,10,11,12,13,14,15);
        // B fragments: lane = col l16; half h -> contiguous K {h*16 .. h*16+15}
        v16bf b0 = *(const v16bf*)(pb0 + kk + h * 16);
        v16bf b1 = *(const v16bf*)(pb1 + kk + h * 16);

        acc00 = __builtin_amdgcn_wmma_f32_16x16x32_bf16(false, a0, false, b0, (short)0, acc00, false, false);
        acc01 = __builtin_amdgcn_wmma_f32_16x16x32_bf16(false, a0, false, b1, (short)0, acc01, false, false);
        acc10 = __builtin_amdgcn_wmma_f32_16x16x32_bf16(false, a1, false, b0, (short)0, acc10, false, false);
        acc11 = __builtin_amdgcn_wmma_f32_16x16x32_bf16(false, a1, false, b1, (short)0, acc11, false, false);
    }

    store_tile(out, acc00, m0,      n0,      l16, h, Nrows, bias, rowscale);
    store_tile(out, acc01, m0,      n0 + 16, l16, h, Nrows, bias, rowscale);
    store_tile(out, acc10, m0 + 16, n0,      l16, h, Nrows, bias, rowscale);
    store_tile(out, acc11, m0 + 16, n0 + 16, l16, h, Nrows, bias, rowscale);
}

// ---------------------------------------------------------------------------
// Per-row softmax entropy + argmax (first-max tie break)
// ent = m + log(S) - T/S  with  S = sum e^(l-m),  T = sum l*e^(l-m)
// ---------------------------------------------------------------------------
__global__ void __launch_bounds__(256)
entropy_argmax_kernel(const float* __restrict__ logits,
                      float* __restrict__ ent, int* __restrict__ yhat, int Kn) {
    __shared__ float sv[256];
    __shared__ int   si[256];
    __shared__ float s1[256];
    __shared__ float s2[256];
    const int row = blockIdx.x;
    const int t   = threadIdx.x;
    const float* lp = logits + (size_t)row * Kn;

    float m = -INFINITY; int mi = 0x7fffffff;
    for (int j = t; j < Kn; j += 256) {
        float v = lp[j];
        if (v > m) { m = v; mi = j; }
    }
    sv[t] = m; si[t] = mi; __syncthreads();
    for (int k = 128; k > 0; k >>= 1) {
        if (t < k) {
            float o = sv[t + k]; int oi = si[t + k];
            if (o > sv[t] || (o == sv[t] && oi < si[t])) { sv[t] = o; si[t] = oi; }
        }
        __syncthreads();
    }
    m = sv[0];
    const int amax = si[0];
    __syncthreads();

    float S = 0.f, T = 0.f;
    for (int j = t; j < Kn; j += 256) {
        float l = lp[j];
        float e = __expf(l - m);
        S += e; T += e * l;
    }
    s1[t] = S; s2[t] = T; __syncthreads();
    for (int k = 128; k > 0; k >>= 1) {
        if (t < k) { s1[t] += s1[t + k]; s2[t] += s2[t + k]; }
        __syncthreads();
    }
    if (t == 0) {
        S = s1[0]; T = s2[0];
        ent[row]  = m + logf(S) - T / S;
        yhat[row] = amax;
    }
}

// ---------------------------------------------------------------------------
// Per-class top-29 lowest-entropy kept samples (serial insertion scan in
// sample order -> matches reference tie-break by earliest slot)
// ---------------------------------------------------------------------------
__global__ void __launch_bounds__(128)
select_topm_kernel(const float* __restrict__ ent, const int* __restrict__ yhat,
                   int Bn, int Kn, int* __restrict__ selcnt, int* __restrict__ selid) {
    const int k = blockIdx.x * 128 + threadIdx.x;
    if (k >= Kn) return;
    float e[MSEL]; int id[MSEL]; int cnt = 0;
    for (int i = 0; i < Bn; ++i) {
        if (yhat[i] != k) continue;
        const float ev = ent[i];
        if (!(ev <= ENT_THR)) continue;
        if (cnt < MSEL) {
            int j = cnt++;
            while (j > 0 && e[j - 1] > ev) { e[j] = e[j - 1]; id[j] = id[j - 1]; --j; }
            e[j] = ev; id[j] = i;
        } else if (ev < e[MSEL - 1]) {
            int j = MSEL - 1;
            while (j > 0 && e[j - 1] > ev) { e[j] = e[j - 1]; id[j] = id[j - 1]; --j; }
            e[j] = ev; id[j] = i;
        }
    }
    selcnt[k] = cnt;
    for (int j = 0; j < cnt; ++j) selid[(size_t)k * MSEL + j] = id[j];
}

// ---------------------------------------------------------------------------
// Per-class centroid of (prototype + selected samples), normalized.
// One 256-thread block per class; D <= 768 assumed (3 elems/thread).
// ---------------------------------------------------------------------------
__global__ void __launch_bounds__(256)
centroid_kernel(const float* __restrict__ W, const float* __restrict__ rw,
                const float* __restrict__ z, const float* __restrict__ rz,
                const int* __restrict__ selcnt, const int* __restrict__ selid,
                float* __restrict__ Cn, int D) {
    __shared__ float sh[256];
    const int k = blockIdx.x;
    const int t = threadIdx.x;
    const float ws = rw[k];
    const float* wr = W + (size_t)k * D;

    float a[3];
    #pragma unroll
    for (int p = 0; p < 3; ++p) {
        const int d = t + p * 256;
        a[p] = (d < D) ? wr[d] * ws : 0.f;           // Wn[k] prototype
    }
    const int cnt = selcnt[k];
    for (int i = 0; i < cnt; ++i) {
        const int   sid = selid[(size_t)k * MSEL + i];
        const float rs  = rz[sid];
        const float* zr = z + (size_t)sid * D;
        #pragma unroll
        for (int p = 0; p < 3; ++p) {
            const int d = t + p * 256;
            if (d < D) a[p] += zr[d] * rs;           // zn[sid]
        }
    }
    const float inv = 1.f / (float)(cnt + 1);
    float sq = 0.f;
    #pragma unroll
    for (int p = 0; p < 3; ++p) { a[p] *= inv; sq += a[p] * a[p]; }

    sh[t] = sq; __syncthreads();
    for (int s = 128; s > 0; s >>= 1) {
        if (t < s) sh[t] += sh[t + s];
        __syncthreads();
    }
    const float rn = 1.f / fmaxf(sqrtf(sh[0]), EPSN);
    float* cr = Cn + (size_t)k * D;
    #pragma unroll
    for (int p = 0; p < 3; ++p) {
        const int d = t + p * 256;
        if (d < D) cr[d] = a[p] * rn;
    }
}

// ---------------------------------------------------------------------------
// Host-side orchestration
// ---------------------------------------------------------------------------
extern "C" void kernel_launch(void* const* d_in, const int* in_sizes, int n_in,
                              void* d_out, int out_size, void* d_ws, size_t ws_size,
                              hipStream_t stream) {
    const float* z = (const float*)d_in[0];   // (B, D)
    const float* W = (const float*)d_in[1];   // (K, D)
    const float* b = (const float*)d_in[2];   // (K,)

    const int Kn = in_sizes[2];
    const int Dn = in_sizes[1] / Kn;
    const int Bn = in_sizes[0] / Dn;

    // workspace carve-out (4-byte units; everything fully written before read)
    float* ws     = (float*)d_ws;
    size_t off    = 0;
    float* logits = ws + off; off += (size_t)Bn * Kn;
    float* ent    = ws + off; off += Bn;
    float* rz     = ws + off; off += Bn;
    float* rw     = ws + off; off += Kn;
    float* Cn     = ws + off; off += (size_t)Kn * Dn;
    int*    yhat   = (int*)(ws + off); off += Bn;
    int*    selcnt = (int*)(ws + off); off += Kn;
    int*    selid  = (int*)(ws + off); off += (size_t)Kn * MSEL;
    bf16_t* zbf    = (bf16_t*)(ws + off); off += (size_t)Bn * Dn / 2;
    bf16_t* Wbf    = (bf16_t*)(ws + off); off += (size_t)Kn * Dn / 2;
    bf16_t* Cnbf   = (bf16_t*)(ws + off); off += (size_t)Kn * Dn / 2;

    const int ntilesN32 = (Kn + 31) / 32;
    const dim3 gemmGrid((ntilesN32 + 3) / 4, Bn / 32);
    const dim3 gemmBlock(128);

    // 1) inverse row norms
    rownorm_inv_kernel<<<Bn, 256, 0, stream>>>(z, rz, Dn);
    rownorm_inv_kernel<<<Kn, 256, 0, stream>>>(W, rw, Dn);

    // 2) one-time bf16 conversions (streaming)
    {
        long zn4 = (long)Bn * Dn / 4;
        long wn4 = (long)Kn * Dn / 4;
        f32_to_bf16_kernel<<<(int)((zn4 + 255) / 256), 256, 0, stream>>>(z, zbf, zn4);
        f32_to_bf16_kernel<<<(int)((wn4 + 255) / 256), 256, 0, stream>>>(W, Wbf, wn4);
    }

    // 3) logits = z @ W^T + b   (bf16 WMMA, fp32 accumulate)
    wmma_gemm_nt_bf16_kernel<<<gemmGrid, gemmBlock, 0, stream>>>(
        zbf, Wbf, b, nullptr, logits, Kn, Dn, ntilesN32);

    // 4) per-row entropy + argmax
    entropy_argmax_kernel<<<Bn, 256, 0, stream>>>(logits, ent, yhat, Kn);

    // 5) per-class top-29 lowest-entropy selection
    select_topm_kernel<<<(Kn + 127) / 128, 128, 0, stream>>>(
        ent, yhat, Bn, Kn, selcnt, selid);

    // 6) per-class normalized centroid Cn, then bf16 conversion
    centroid_kernel<<<Kn, 256, 0, stream>>>(W, rw, z, rz, selcnt, selid, Cn, Dn);
    {
        long cn4 = (long)Kn * Dn / 4;
        f32_to_bf16_kernel<<<(int)((cn4 + 255) / 256), 256, 0, stream>>>(Cn, Cnbf, cn4);
    }

    // 7) out = zn @ Cn^T  ==  rowscale(rz) * (z @ Cn^T)
    wmma_gemm_nt_bf16_kernel<<<gemmGrid, gemmBlock, 0, stream>>>(
        zbf, Cnbf, nullptr, rz, (float*)d_out, Kn, Dn, ntilesN32);
}
